// GNNTransformerDecoderLayer_66529043415208
// MI455X (gfx1250) — matrix operand order repaired and checked by reading, not compiled, not run on hardware
//
#include <hip/hip_runtime.h>

// Problem constants (match reference)
#define BB  8
#define TT  512
#define SSZ 512
#define DD  512
#define HH  8
#define HDD 64
#define LLE 8
#define FF  2048

typedef __bf16 bf16;
typedef __attribute__((ext_vector_type(16))) __bf16 v16bf;
typedef __attribute__((ext_vector_type(8)))  __bf16 bf16x8;
typedef __attribute__((ext_vector_type(8)))  float  v8f;

// ---------------- scalar bf16 helpers ----------------
__device__ __forceinline__ bf16 f2bf(float f) {
  union { float f; unsigned u; } v; v.f = f;
  unsigned r = v.u + 0x7FFFu + ((v.u >> 16) & 1u);
  union { unsigned short s; bf16 b; } o; o.s = (unsigned short)(r >> 16);
  return o.b;
}

// ---------------- WMMA fragment loaders (CDNA5 wave32 layouts) ----------------
// A fragment, 16x32 bf16 (MxK), row-major A with row stride lda, base at (row0,k0).
// lane<16: row=lane, K = {0..15}; lane>=16: row=lane-16, K = {16..31}
__device__ __forceinline__ v16bf load_a_frag(const bf16* __restrict__ A, int lda, int lane) {
  const bf16* p = A + (size_t)(lane & 15) * lda + ((lane >> 4) << 4);
  bf16x8 lo = *(const bf16x8*)(p);
  bf16x8 hi = *(const bf16x8*)(p + 8);
  return __builtin_shufflevector(lo, hi, 0,1,2,3,4,5,6,7,8,9,10,11,12,13,14,15);
}

// B fragment, 32x16 bf16 (KxN) where B[k][n] = W[n][k] (x @ W^T pattern).
// W row-major [N,K], row stride ldw, base at (n0,k0).  16 contiguous k per lane.
__device__ __forceinline__ v16bf load_b_wt(const bf16* __restrict__ W, int ldw, int lane) {
  const bf16* p = W + (size_t)(lane & 15) * ldw + ((lane >> 4) << 4);
  bf16x8 lo = *(const bf16x8*)(p);
  bf16x8 hi = *(const bf16x8*)(p + 8);
  return __builtin_shufflevector(lo, hi, 0,1,2,3,4,5,6,7,8,9,10,11,12,13,14,15);
}

// ---------------- block reductions (LDS) ----------------
__device__ __forceinline__ float block_sum(float v, float* sm) {
  int tid = threadIdx.x;
  sm[tid] = v; __syncthreads();
  for (int o = 128; o > 0; o >>= 1) { if (tid < o) sm[tid] += sm[tid + o]; __syncthreads(); }
  float r = sm[0]; __syncthreads();
  return r;
}
__device__ __forceinline__ float block_max(float v, float* sm) {
  int tid = threadIdx.x;
  sm[tid] = v; __syncthreads();
  for (int o = 128; o > 0; o >>= 1) { if (tid < o) sm[tid] = fmaxf(sm[tid], sm[tid + o]); __syncthreads(); }
  float r = sm[0]; __syncthreads();
  return r;
}

// ---------------- elementwise conversion kernels ----------------
__global__ void cvt_kernel(const float* __restrict__ src, bf16* __restrict__ dst, long n) {
  long i = (long)blockIdx.x * blockDim.x + threadIdx.x;
  if (i < n) dst[i] = f2bf(src[i]);
}

// src [Trows,B,D] f32 -> dst [B,Trows,D] bf16
__global__ void xpose_kernel(const float* __restrict__ src, bf16* __restrict__ dst, int Trows) {
  long i = (long)blockIdx.x * blockDim.x + threadIdx.x;
  long n = (long)Trows * BB * DD;
  if (i >= n) return;
  int d = (int)(i % DD);
  int t = (int)((i / DD) % Trows);
  int b = (int)(i / ((long)DD * Trows));
  dst[((size_t)b * Trows + t) * DD + d] = f2bf(src[((size_t)t * BB + b) * DD + d]);
}

// rel [L,HD,HD] f32 -> relT[l][e][d] = rel[l][d][e] bf16
__global__ void rel_xpose_kernel(const float* __restrict__ rel, bf16* __restrict__ relT) {
  int i = blockIdx.x * blockDim.x + threadIdx.x;
  if (i >= LLE * HDD * HDD) return;
  int e = i % HDD;
  int d = (i / HDD) % HDD;
  int l = i / (HDD * HDD);
  relT[((size_t)l * HDD + e) * HDD + d] = f2bf(rel[((size_t)l * HDD + d) * HDD + e]);
}

// V [B*Sc, D] bf16 -> VT [B,H,HD,Sc] bf16  (VT[b,h,d,s] = V[b*Sc+s, h*HD+d])
__global__ void vt_kernel(const bf16* __restrict__ V, bf16* __restrict__ VT, int Sc) {
  long i = (long)blockIdx.x * blockDim.x + threadIdx.x;
  long n = (long)BB * HH * HDD * Sc;
  if (i >= n) return;
  int s  = (int)(i % Sc);
  int d  = (int)((i / Sc) % HDD);
  int h  = (int)((i / ((long)Sc * HDD)) % HH);
  int b  = (int)(i / ((long)Sc * HDD * HH));
  VT[i] = V[((size_t)b * Sc + s) * DD + h * HDD + d];
}

// ------- generic GEMM: C[M,N] = A[M,K] @ W[N,K]^T + bias ; wave owns 32x64 tile -------
template <bool BF16OUT, bool RELU>
__global__ void gemm_xwt_kernel(const bf16* __restrict__ A, const bf16* __restrict__ W,
                                const float* __restrict__ bias, void* __restrict__ Cout,
                                int M, int N, int K) {
  int lane = threadIdx.x & 31;
  int wave = blockIdx.x * (blockDim.x >> 5) + (threadIdx.x >> 5);
  int ntiles = N >> 6;                 // 64-wide tiles
  int mt = wave / ntiles, nt = wave % ntiles;
  if (mt >= (M >> 5)) return;
  const bf16* Ab = A + (size_t)(mt * 32) * K;
  const bf16* Wb = W + (size_t)(nt * 64) * K;
  v8f acc[2][4];
#pragma unroll
  for (int mi = 0; mi < 2; ++mi)
#pragma unroll
    for (int j = 0; j < 4; ++j) acc[mi][j] = (v8f){};
  for (int k0 = 0; k0 < K; k0 += 32) {
    __builtin_prefetch(Ab + k0 + 256, 0, 1);
    __builtin_prefetch(Wb + k0 + 256, 0, 1);
    v16bf a0 = load_a_frag(Ab + k0, K, lane);
    v16bf a1 = load_a_frag(Ab + (size_t)16 * K + k0, K, lane);
#pragma unroll
    for (int j = 0; j < 4; ++j) {
      v16bf b = load_b_wt(Wb + (size_t)(j * 16) * K + k0, K, lane);
      acc[0][j] = __builtin_amdgcn_wmma_f32_16x16x32_bf16(false, a0, false, b, (short)0, acc[0][j], false, false);
      acc[1][j] = __builtin_amdgcn_wmma_f32_16x16x32_bf16(false, a1, false, b, (short)0, acc[1][j], false, false);
    }
  }
  int lc = lane & 15;
  int r0 = (lane >> 4) << 3;
  float bv[4];
#pragma unroll
  for (int j = 0; j < 4; ++j) bv[j] = bias[nt * 64 + j * 16 + lc];
#pragma unroll
  for (int mi = 0; mi < 2; ++mi) {
    int rbase = mt * 32 + mi * 16 + r0;
#pragma unroll
    for (int j = 0; j < 4; ++j) {
      int col = nt * 64 + j * 16 + lc;
#pragma unroll
      for (int i = 0; i < 8; ++i) {
        float v = acc[mi][j][i] + bv[j];
        if (RELU) v = v > 0.f ? v : 0.f;
        if (BF16OUT) ((bf16*)Cout)[(size_t)(rbase + i) * N + col] = f2bf(v);
        else         ((float*)Cout)[(size_t)(rbase + i) * N + col] = v;
      }
    }
  }
}

// ------- kl[l,b,h] = K[b,h] @ relT[l]^T ; wave owns 16x64 (full HD) -------
__global__ void gemm_kl_kernel(const bf16* __restrict__ Kmat, const bf16* __restrict__ relT,
                               bf16* __restrict__ kl) {
  int lane = threadIdx.x & 31;
  int mt = blockIdx.x * 8 + (threadIdx.x >> 5);   // 0..31 (T/16)
  int batch = blockIdx.y;                          // l*B*H + b*H + h
  int h = batch & 7, b = (batch >> 3) & 7, l = batch >> 6;
  const bf16* Ab = Kmat + ((size_t)(b * TT + mt * 16)) * DD + h * HDD;
  const bf16* Wb = relT + (size_t)l * HDD * HDD;
  v8f acc[4];
#pragma unroll
  for (int j = 0; j < 4; ++j) acc[j] = (v8f){};
#pragma unroll
  for (int k0 = 0; k0 < HDD; k0 += 32) {
    v16bf a = load_a_frag(Ab + k0, DD, lane);
#pragma unroll
    for (int j = 0; j < 4; ++j) {
      v16bf bfr = load_b_wt(Wb + (size_t)(j * 16) * HDD + k0, HDD, lane);
      acc[j] = __builtin_amdgcn_wmma_f32_16x16x32_bf16(false, a, false, bfr, (short)0, acc[j], false, false);
    }
  }
  bf16* Cb = kl + (size_t)batch * TT * HDD + (size_t)(mt * 16) * HDD;
  int lc = lane & 15;
  int r0 = (lane >> 4) << 3;
#pragma unroll
  for (int j = 0; j < 4; ++j)
#pragma unroll
    for (int i = 0; i < 8; ++i)
      Cb[(size_t)(r0 + i) * HDD + j * 16 + lc] = f2bf(acc[j][i]);
}

// -------- scores[b,h,i,j] = scale * sum_l em[b,l,i,j] * (Q[i,:] . kl[l,b,h,j,:]) --------
__global__ void sa_scores_kernel(const bf16* __restrict__ Q, const bf16* __restrict__ kl,
                                 const float* __restrict__ em, const unsigned char* __restrict__ pad,
                                 float* __restrict__ scores) {
  int lane = threadIdx.x & 31;
  int wave = blockIdx.x * 8 + (threadIdx.x >> 5);   // 0..1023
  int bh = blockIdx.y;
  int b = bh >> 3, h = bh & 7;
  int mt = wave >> 5;   // 32 i-tiles
  int nt = wave & 31;   // 32 j-tiles
  int col = nt * 16 + (lane & 15);
  int r0  = mt * 16 + ((lane >> 4) << 3);
  const bf16* Ab = Q + ((size_t)(b * TT + mt * 16)) * DD + h * HDD;
  // Q fragments are l-invariant: load once
  v16bf a0 = load_a_frag(Ab + 0,  DD, lane);
  v16bf a1 = load_a_frag(Ab + 32, DD, lane);
  float tot[8] = {0.f, 0.f, 0.f, 0.f, 0.f, 0.f, 0.f, 0.f};
#pragma unroll
  for (int l = 0; l < LLE; ++l) {
    const bf16* Wb = kl + (((size_t)l * BB + b) * HH + h) * TT * HDD + (size_t)(nt * 16) * HDD;
    v16bf b0 = load_b_wt(Wb + 0,  HDD, lane);
    v16bf b1 = load_b_wt(Wb + 32, HDD, lane);
    v8f acc = {};
    acc = __builtin_amdgcn_wmma_f32_16x16x32_bf16(false, a0, false, b0, (short)0, acc, false, false);
    acc = __builtin_amdgcn_wmma_f32_16x16x32_bf16(false, a1, false, b1, (short)0, acc, false, false);
    const float* emp = em + ((size_t)b * LLE + l) * TT * TT;
#pragma unroll
    for (int i = 0; i < 8; ++i) tot[i] += emp[(size_t)(r0 + i) * TT + col] * acc[i];
  }
  const unsigned char* pp = pad + (size_t)b * TT * TT;
  float* sp = scores + ((size_t)bh * TT) * TT;
#pragma unroll
  for (int i = 0; i < 8; ++i) {
    float v = tot[i] * 0.125f;                       // 1/sqrt(64)
    if (pp[(size_t)(r0 + i) * TT + col]) v = -1.0e9f;
    sp[(size_t)(r0 + i) * TT + col] = v;
  }
}

// -------- cross scores: cs[b,h,t,s] = scale * CQ . CK ; wave owns 16x64 --------
__global__ void ca_scores_kernel(const bf16* __restrict__ CQ, const bf16* __restrict__ CK,
                                 float* __restrict__ cs) {
  int lane = threadIdx.x & 31;
  int wave = blockIdx.x * 8 + (threadIdx.x >> 5);   // 0..255
  int bh = blockIdx.y;
  int b = bh >> 3, h = bh & 7;
  int mt = wave >> 3;   // 32 t-tiles
  int nt = wave & 7;    // 8 s-tiles of 64
  const bf16* Ab = CQ + ((size_t)(b * TT + mt * 16)) * DD + h * HDD;
  v16bf a0 = load_a_frag(Ab + 0,  DD, lane);
  v16bf a1 = load_a_frag(Ab + 32, DD, lane);
  v8f acc[4];
#pragma unroll
  for (int j = 0; j < 4; ++j) acc[j] = (v8f){};
#pragma unroll
  for (int j = 0; j < 4; ++j) {
    const bf16* Wb = CK + ((size_t)(b * SSZ + nt * 64 + j * 16)) * DD + h * HDD;
    v16bf b0 = load_b_wt(Wb + 0,  DD, lane);
    v16bf b1 = load_b_wt(Wb + 32, DD, lane);
    acc[j] = __builtin_amdgcn_wmma_f32_16x16x32_bf16(false, a0, false, b0, (short)0, acc[j], false, false);
    acc[j] = __builtin_amdgcn_wmma_f32_16x16x32_bf16(false, a1, false, b1, (short)0, acc[j], false, false);
  }
  int lc = lane & 15;
  int r0 = mt * 16 + ((lane >> 4) << 3);
  float* sp = cs + ((size_t)bh * TT) * SSZ;
#pragma unroll
  for (int j = 0; j < 4; ++j)
#pragma unroll
    for (int i = 0; i < 8; ++i)
      sp[(size_t)(r0 + i) * SSZ + nt * 64 + j * 16 + lc] = acc[j][i] * 0.125f;
}

// ---------------- row softmax (512 cols), f32 -> bf16 ----------------
__global__ void softmax_kernel(const float* __restrict__ src, bf16* __restrict__ dst) {
  __shared__ float sm[256];
  size_t row = blockIdx.x;
  int tid = threadIdx.x;
  const float* r = src + row * 512;
  float a = r[tid], b = r[tid + 256];
  float m = block_max(fmaxf(a, b), sm);
  float ea = __expf(a - m), eb = __expf(b - m);
  float s = block_sum(ea + eb, sm);
  float inv = 1.f / s;
  bf16* o = dst + row * 512;
  o[tid] = f2bf(ea * inv);
  o[tid + 256] = f2bf(eb * inv);
}

// ------- O[b,t,h*HD+d] = attn[b,h] @ V[b,h] using VT[b,h,d,s]; wave owns 16x64 -------
__global__ void attn_v_kernel(const bf16* __restrict__ attn, const bf16* __restrict__ VT,
                              bf16* __restrict__ O, int Sc) {
  int lane = threadIdx.x & 31;
  int mt = blockIdx.x * 8 + (threadIdx.x >> 5);   // 0..31 t-tiles
  int bh = blockIdx.y;
  int b = bh >> 3, h = bh & 7;
  const bf16* Ab = attn + ((size_t)bh * TT + mt * 16) * Sc;
  const bf16* Wb = VT + (size_t)bh * HDD * Sc;     // rows = d (N), ld = Sc
  v8f acc[4];
#pragma unroll
  for (int j = 0; j < 4; ++j) acc[j] = (v8f){};
  for (int k0 = 0; k0 < Sc; k0 += 32) {
    __builtin_prefetch(Ab + k0 + 256, 0, 1);
    v16bf a = load_a_frag(Ab + k0, Sc, lane);
#pragma unroll
    for (int j = 0; j < 4; ++j) {
      v16bf bfr = load_b_wt(Wb + (size_t)(j * 16) * Sc + k0, Sc, lane);
      acc[j] = __builtin_amdgcn_wmma_f32_16x16x32_bf16(false, a, false, bfr, (short)0, acc[j], false, false);
    }
  }
  bf16* Ob = O + ((size_t)(b * TT + mt * 16)) * DD + h * HDD;
  int lc = lane & 15;
  int r0 = (lane >> 4) << 3;
#pragma unroll
  for (int j = 0; j < 4; ++j)
#pragma unroll
    for (int i = 0; i < 8; ++i)
      Ob[(size_t)(r0 + i) * DD + j * 16 + lc] = f2bf(acc[j][i]);
}

// ---------------- LayerNorm with residual; optional f32/bf16/[T,B,D] outputs ----------------
__global__ void ln_kernel(const float* __restrict__ y, const float* __restrict__ res, int resTBD,
                          const float* __restrict__ s, const float* __restrict__ bi,
                          float* __restrict__ outF, bf16* __restrict__ outB,
                          float* __restrict__ outTBD) {
  __shared__ float sm[256];
  int row = blockIdx.x;          // b*T + t
  int b = row / TT, t = row % TT;
  int tid = threadIdx.x;
  int d0 = tid, d1 = tid + 256;
  float r0 = resTBD ? res[((size_t)t * BB + b) * DD + d0] : res[(size_t)row * DD + d0];
  float r1 = resTBD ? res[((size_t)t * BB + b) * DD + d1] : res[(size_t)row * DD + d1];
  float x0 = y[(size_t)row * DD + d0] + r0;
  float x1 = y[(size_t)row * DD + d1] + r1;
  float mean = block_sum(x0 + x1, sm) * (1.f / DD);
  float e0 = x0 - mean, e1 = x1 - mean;
  float var = block_sum(e0 * e0 + e1 * e1, sm) * (1.f / DD);
  float inv = rsqrtf(var + 1e-5f);
  float v0 = e0 * inv * s[d0] + bi[d0];
  float v1 = e1 * inv * s[d1] + bi[d1];
  if (outF)   { outF[(size_t)row * DD + d0] = v0; outF[(size_t)row * DD + d1] = v1; }
  if (outB)   { outB[(size_t)row * DD + d0] = f2bf(v0); outB[(size_t)row * DD + d1] = f2bf(v1); }
  if (outTBD) { outTBD[((size_t)t * BB + b) * DD + d0] = v0;
                outTBD[((size_t)t * BB + b) * DD + d1] = v1; }
}

// ==================================================================================
extern "C" void kernel_launch(void* const* d_in, const int* in_sizes, int n_in,
                              void* d_out, int out_size, void* d_ws, size_t ws_size,
                              hipStream_t stream) {
  (void)in_sizes; (void)n_in; (void)out_size; (void)ws_size;
  const float* tgt       = (const float*)d_in[0];
  const float* edge_mask = (const float*)d_in[1];
  const unsigned char* pad = (const unsigned char*)d_in[2];
  const float* memory    = (const float*)d_in[3];
  const float* sa_q_w = (const float*)d_in[4];  const float* sa_q_b = (const float*)d_in[5];
  const float* sa_k_w = (const float*)d_in[6];  const float* sa_k_b = (const float*)d_in[7];
  const float* sa_v_w = (const float*)d_in[8];  const float* sa_v_b = (const float*)d_in[9];
  const float* sa_rel = (const float*)d_in[10];
  const float* sa_out_w = (const float*)d_in[11]; const float* sa_out_b = (const float*)d_in[12];
  const float* ca_in_w = (const float*)d_in[13];  const float* ca_in_b = (const float*)d_in[14];
  const float* ca_out_w = (const float*)d_in[15]; const float* ca_out_b = (const float*)d_in[16];
  const float* lin1_w = (const float*)d_in[17];   const float* lin1_b = (const float*)d_in[18];
  const float* lin2_w = (const float*)d_in[19];   const float* lin2_b = (const float*)d_in[20];
  const float* ln1_s = (const float*)d_in[21];    const float* ln1_b = (const float*)d_in[22];
  const float* ln2_s = (const float*)d_in[23];    const float* ln2_b = (const float*)d_in[24];
  const float* ln3_s = (const float*)d_in[25];    const float* ln3_b = (const float*)d_in[26];
  float* out = (float*)d_out;

  char* ws = (char*)d_ws;
  size_t off = 0;
  auto alloc = [&](size_t bytes) -> char* {
    char* p = ws + off;
    off = (off + bytes + 255) & ~(size_t)255;
    return p;
  };

  bf16* xb    = (bf16*)alloc((size_t)BB * TT * DD * 2);
  bf16* memb  = (bf16*)alloc((size_t)BB * SSZ * DD * 2);
  bf16* wq    = (bf16*)alloc((size_t)DD * DD * 2);
  bf16* wk    = (bf16*)alloc((size_t)DD * DD * 2);
  bf16* wv    = (bf16*)alloc((size_t)DD * DD * 2);
  bf16* wo    = (bf16*)alloc((size_t)DD * DD * 2);
  bf16* relT  = (bf16*)alloc((size_t)LLE * HDD * HDD * 2);
  bf16* wca   = (bf16*)alloc((size_t)3 * DD * DD * 2);
  bf16* wcao  = (bf16*)alloc((size_t)DD * DD * 2);
  bf16* wl1   = (bf16*)alloc((size_t)FF * DD * 2);
  bf16* wl2   = (bf16*)alloc((size_t)DD * FF * 2);
  bf16* Qb    = (bf16*)alloc((size_t)BB * TT * DD * 2);   // reused as CQ
  bf16* Kb    = (bf16*)alloc((size_t)BB * TT * DD * 2);   // reused as CK
  bf16* Vb    = (bf16*)alloc((size_t)BB * TT * DD * 2);   // reused as CV
  bf16* VT    = (bf16*)alloc((size_t)BB * TT * DD * 2);   // V transposed, reused for cross
  bf16* kl    = (bf16*)alloc((size_t)LLE * BB * HH * TT * HDD * 2); // reused as FFN hidden
  float* scores = (float*)alloc((size_t)BB * HH * TT * TT * 4);     // reused for cross
  bf16* attn  = (bf16*)alloc((size_t)BB * HH * TT * TT * 2);        // reused for cross
  bf16* Ob    = (bf16*)alloc((size_t)BB * TT * DD * 2);             // reused as CO
  float* ybuf = (float*)alloc((size_t)BB * TT * DD * 4);            // reused for y2/y3
  float* x1f  = (float*)alloc((size_t)BB * TT * DD * 4);
  bf16* x1b   = (bf16*)alloc((size_t)BB * TT * DD * 2);
  float* x2f  = (float*)alloc((size_t)BB * TT * DD * 4);
  bf16* x2b   = (bf16*)alloc((size_t)BB * TT * DD * 2);
  bf16* h1    = kl;   // FFN hidden reuses kl region (16.8MB <= 33.5MB)

  const int THR = 256;
  const int M = BB * TT;                               // 4096
  const int GEMM_B_512  = (M / 32) * (DD / 64) / 8;    // 128 blocks
  const int GEMM_B_2048 = (M / 32) * (FF / 64) / 8;    // 512 blocks

  // ---- converts / transposes ----
  xpose_kernel<<<(BB * TT * DD) / THR, THR, 0, stream>>>(tgt, xb, TT);
  xpose_kernel<<<(BB * SSZ * DD) / THR, THR, 0, stream>>>(memory, memb, SSZ);
  cvt_kernel<<<(DD * DD) / THR, THR, 0, stream>>>(sa_q_w, wq, (long)DD * DD);
  cvt_kernel<<<(DD * DD) / THR, THR, 0, stream>>>(sa_k_w, wk, (long)DD * DD);
  cvt_kernel<<<(DD * DD) / THR, THR, 0, stream>>>(sa_v_w, wv, (long)DD * DD);
  cvt_kernel<<<(DD * DD) / THR, THR, 0, stream>>>(sa_out_w, wo, (long)DD * DD);
  cvt_kernel<<<(3 * DD * DD) / THR, THR, 0, stream>>>(ca_in_w, wca, (long)3 * DD * DD);
  cvt_kernel<<<(DD * DD) / THR, THR, 0, stream>>>(ca_out_w, wcao, (long)DD * DD);
  cvt_kernel<<<(FF * DD) / THR, THR, 0, stream>>>(lin1_w, wl1, (long)FF * DD);
  cvt_kernel<<<(DD * FF) / THR, THR, 0, stream>>>(lin2_w, wl2, (long)DD * FF);
  rel_xpose_kernel<<<(LLE * HDD * HDD) / THR, THR, 0, stream>>>(sa_rel, relT);

  // ---- self-attention ----
  gemm_xwt_kernel<true, false><<<GEMM_B_512, THR, 0, stream>>>(xb, wq, sa_q_b, Qb, M, DD, DD);
  gemm_xwt_kernel<true, false><<<GEMM_B_512, THR, 0, stream>>>(xb, wk, sa_k_b, Kb, M, DD, DD);
  gemm_xwt_kernel<true, false><<<GEMM_B_512, THR, 0, stream>>>(xb, wv, sa_v_b, Vb, M, DD, DD);
  gemm_kl_kernel<<<dim3(4, LLE * BB * HH), THR, 0, stream>>>(Kb, relT, kl);
  sa_scores_kernel<<<dim3(128, BB * HH), THR, 0, stream>>>(Qb, kl, edge_mask, pad, scores);
  softmax_kernel<<<BB * HH * TT, THR, 0, stream>>>(scores, attn);
  vt_kernel<<<(BB * TT * DD) / THR, THR, 0, stream>>>(Vb, VT, TT);
  attn_v_kernel<<<dim3(4, BB * HH), THR, 0, stream>>>(attn, VT, Ob, TT);
  gemm_xwt_kernel<false, false><<<GEMM_B_512, THR, 0, stream>>>(Ob, wo, sa_out_b, ybuf, M, DD, DD);
  ln_kernel<<<M, THR, 0, stream>>>(ybuf, tgt, 1, ln1_s, ln1_b, x1f, x1b, nullptr);

  // ---- cross-attention ----
  gemm_xwt_kernel<true, false><<<GEMM_B_512, THR, 0, stream>>>(x1b, wca, ca_in_b, Qb, M, DD, DD);
  gemm_xwt_kernel<true, false><<<GEMM_B_512, THR, 0, stream>>>(memb, wca + (size_t)DD * DD, ca_in_b + DD, Kb, BB * SSZ, DD, DD);
  gemm_xwt_kernel<true, false><<<GEMM_B_512, THR, 0, stream>>>(memb, wca + (size_t)2 * DD * DD, ca_in_b + 2 * DD, Vb, BB * SSZ, DD, DD);
  ca_scores_kernel<<<dim3(32, BB * HH), THR, 0, stream>>>(Qb, Kb, scores);
  softmax_kernel<<<BB * HH * TT, THR, 0, stream>>>(scores, attn);
  vt_kernel<<<(BB * SSZ * DD) / THR, THR, 0, stream>>>(Vb, VT, SSZ);
  attn_v_kernel<<<dim3(4, BB * HH), THR, 0, stream>>>(attn, VT, Ob, SSZ);
  gemm_xwt_kernel<false, false><<<GEMM_B_512, THR, 0, stream>>>(Ob, wcao, ca_out_b, ybuf, M, DD, DD);
  ln_kernel<<<M, THR, 0, stream>>>(ybuf, x1f, 0, ln2_s, ln2_b, x2f, x2b, nullptr);

  // ---- FFN ----
  gemm_xwt_kernel<true, true><<<GEMM_B_2048, THR, 0, stream>>>(x2b, wl1, lin1_b, h1, M, FF, DD);
  gemm_xwt_kernel<false, false><<<GEMM_B_512, THR, 0, stream>>>(h1, wl2, lin2_b, ybuf, M, DD, FF);
  ln_kernel<<<M, THR, 0, stream>>>(ybuf, x2f, 0, ln3_s, ln3_b, nullptr, nullptr, out);
}
